// GeneGraphEncoder_64355789963645
// MI455X (gfx1250) — matrix operand order repaired and checked by reading, not compiled, not run on hardware
//
#include <hip/hip_runtime.h>
#include <hip/hip_bf16.h>

// ---------------------------------------------------------------------------
// GeneGraphEncoder forward for MI455X (gfx1250, wave32, WMMA).
// GEMMs consume f16 operands laid out so WMMA fragments are contiguous
// 16B/32B runs in memory -> direct global_load_b128 fragments, no LDS/barriers
// in the hot loop. Each wave owns two N-tiles (shared A fragment, two
// independent accumulator chains) with a 2-deep software pipeline.
// ---------------------------------------------------------------------------

typedef __attribute__((ext_vector_type(16))) _Float16 v16h;
typedef __attribute__((ext_vector_type(8)))  _Float16 v8h;
typedef __attribute__((ext_vector_type(4)))  _Float16 v4h;
typedef __attribute__((ext_vector_type(8)))  float    v8f;

#define Gn_    768
#define NFD_   64
#define HID_   256
#define NH_    8
#define HD_    32
#define NEF_   16
#define GG_    (Gn_ * Gn_)
#define NX_    (Gn_ * HID_)

__device__ __forceinline__ float eluf(float x) {
  return x > 0.f ? x : (__expf(x) - 1.f);
}

// --------------------------------------------------------------------------
// f32 -> f16 bulk convert (n multiple of 4).
// --------------------------------------------------------------------------
__global__ void cvt_f16_kernel(const float* __restrict__ in,
                               _Float16* __restrict__ out, long long n4)
{
  const long long i = (long long)blockIdx.x * blockDim.x + threadIdx.x;
  if (i >= n4) return;
  const float4 v = ((const float4*)in)[i];
  const v4h p = { (_Float16)v.x, (_Float16)v.y, (_Float16)v.z, (_Float16)v.w };
  ((v4h*)out)[i] = p;
}

// --------------------------------------------------------------------------
// Transpose-convert: in (R x C, f32) -> out (C x R, f16). 32x32 LDS tiles.
// --------------------------------------------------------------------------
__global__ void transpose_f16_kernel(const float* __restrict__ in,
                                     _Float16* __restrict__ out, int R, int C)
{
  __shared__ float tile[32][33];
  const int c0 = blockIdx.x * 32, r0 = blockIdx.y * 32;
  const int tx = threadIdx.x & 31, ty = threadIdx.x >> 5;   // 256 threads
#pragma unroll
  for (int i = 0; i < 32; i += 8) {
    const int r = r0 + ty + i, c = c0 + tx;
    tile[ty + i][tx] = (r < R && c < C) ? in[(size_t)r * C + c] : 0.f;
  }
  __syncthreads();
#pragma unroll
  for (int i = 0; i < 32; i += 8) {
    const int c = c0 + ty + i, r = r0 + tx;
    if (c < C && r < R) out[(size_t)c * R + r] = (_Float16)tile[tx][ty + i];
  }
}

// --------------------------------------------------------------------------
// WMMA GEMM, f16 operands: C[z] = act(alpha*(A[z] @ Bt[z]^T) + bias) + resid
// A: row-major MxK f16.  Bt: row-major NxK f16 (B transposed).
// K%32==0, M%16==0, N%32==0. Wave = 16x32 of C (2 tiles, shared A fragment,
// 2 accumulator chains); block of 4 waves = 16x128. Fragments are direct
// global b128 loads; next K-step is issued before the current WMMAs.
// Optionally writes an f16 copy of C (same ldc) for chained GEMMs.
// --------------------------------------------------------------------------
__global__ void wmma_gemm_f16_kernel(
    const _Float16* __restrict__ A, int lda, long long sAz,
    const _Float16* __restrict__ Bt, int ldbt, long long sBz,
    const float* __restrict__ bias,
    const float* __restrict__ residual,
    float* __restrict__ C, int ldc, long long sCz,
    _Float16* __restrict__ C16, long long sC16z,
    int N, int K, float alpha, int act)
{
  const int wave = threadIdx.x >> 5, lane = threadIdx.x & 31;
  const int m = lane & 15, hs = lane >> 4;
  const int row0 = blockIdx.y * 16;
  const int colW = blockIdx.x * 128 + wave * 32;
  if (colW >= N) return;

  const _Float16* __restrict__ arow =
      A + (size_t)blockIdx.z * sAz + (size_t)(row0 + m) * lda;
  const _Float16* __restrict__ brow0 =
      Bt + (size_t)blockIdx.z * sBz + (size_t)(colW + m) * ldbt;
  const _Float16* __restrict__ brow1 = brow0 + (size_t)16 * ldbt;

  // stage-0 fragments (A: K=hs*16+e ; B: K=hs*16+e)
  v8h a0 = *(const v8h*)&arow[hs * 16];
  v8h a1 = *(const v8h*)&arow[hs * 16 + 8];
  v8h c0 = *(const v8h*)&brow0[hs * 16];
  v8h c1 = *(const v8h*)&brow0[hs * 16 + 8];
  v8h d0 = *(const v8h*)&brow1[hs * 16];
  v8h d1 = *(const v8h*)&brow1[hs * 16 + 8];

  v8f acc0 = {}, acc1 = {};
  for (int k0 = 0; k0 < K; k0 += 32) {
    const int kn = (k0 + 32 < K) ? (k0 + 32) : 0;   // branch-free tail reload
    if (k0 + 64 < K) {
      __builtin_prefetch(&arow[k0 + 64 + hs * 8], 0, 3);
      __builtin_prefetch(&brow0[k0 + 64 + hs * 16], 0, 3);
      __builtin_prefetch(&brow1[k0 + 64 + hs * 16], 0, 3);
    }
    // issue next-step loads before consuming current fragments
    const v8h na0 = *(const v8h*)&arow[kn + hs * 16];
    const v8h na1 = *(const v8h*)&arow[kn + hs * 16 + 8];
    const v8h nc0 = *(const v8h*)&brow0[kn + hs * 16];
    const v8h nc1 = *(const v8h*)&brow0[kn + hs * 16 + 8];
    const v8h nd0 = *(const v8h*)&brow1[kn + hs * 16];
    const v8h nd1 = *(const v8h*)&brow1[kn + hs * 16 + 8];

    v16h a, b0, b1;
#pragma unroll
    for (int e = 0; e < 8; ++e) {
      a[e]  = a0[e]; a[e + 8]  = a1[e];
      b0[e] = c0[e]; b0[e + 8] = c1[e];
      b1[e] = d0[e]; b1[e + 8] = d1[e];
    }
    acc0 = __builtin_amdgcn_wmma_f32_16x16x32_f16(
        false, a, false, b0, (short)0, acc0, false, false);
    acc1 = __builtin_amdgcn_wmma_f32_16x16x32_f16(
        false, a, false, b1, (short)0, acc1, false, false);

    a0 = na0; a1 = na1; c0 = nc0; c1 = nc1; d0 = nd0; d1 = nd1;
  }

#pragma unroll
  for (int tt = 0; tt < 2; ++tt) {
    const v8f acc = tt ? acc1 : acc0;
    const int col = colW + tt * 16 + m;
#pragma unroll
    for (int r = 0; r < 8; ++r) {
      const int row = row0 + r + (hs << 3);   // C layout: VGPR r -> M = r+8*hs
      float val = acc[r] * alpha;
      if (bias)     val += bias[col];
      if (act == 1) val  = eluf(val);
      if (residual) val += residual[(size_t)row * ldc + col];
      C[(size_t)blockIdx.z * sCz + (size_t)row * ldc + col] = val;
      if (C16)
        C16[(size_t)blockIdx.z * sC16z + (size_t)row * ldc + col] =
            (_Float16)val;
    }
  }
}

// --------------------------------------------------------------------------
// Per-row LayerNorm (d == blockDim.x == 256), optional fused elu + f16 copy.
// --------------------------------------------------------------------------
__global__ void layernorm_kernel(const float* __restrict__ x,
                                 const float* __restrict__ g,
                                 const float* __restrict__ b,
                                 float* __restrict__ out,
                                 _Float16* __restrict__ out16,
                                 int d, int act)
{
  const int row = blockIdx.x;
  const int tid = threadIdx.x;
  const float val = x[(size_t)row * d + tid];
  __shared__ float red[8];

  float s = val;
  for (int off = 16; off; off >>= 1) s += __shfl_xor(s, off, 32);
  if ((tid & 31) == 0) red[tid >> 5] = s;
  __syncthreads();
  float tot = 0.f;
#pragma unroll
  for (int w = 0; w < 8; ++w) tot += red[w];
  const float mean = tot / (float)d;
  __syncthreads();

  const float dv = val - mean;
  s = dv * dv;
  for (int off = 16; off; off >>= 1) s += __shfl_xor(s, off, 32);
  if ((tid & 31) == 0) red[tid >> 5] = s;
  __syncthreads();
  tot = 0.f;
#pragma unroll
  for (int w = 0; w < 8; ++w) tot += red[w];
  const float var = tot / (float)d;

  float o = dv * __frsqrt_rn(var + 1e-5f) * g[tid] + b[tid];
  if (act) o = eluf(o);
  out[(size_t)row * d + tid] = o;
  if (out16) out16[(size_t)row * d + tid] = (_Float16)o;
}

// --------------------------------------------------------------------------
// Edge-bias MLP (16 -> elu(16) -> 8) in registers, accumulated into logits.
// --------------------------------------------------------------------------
__global__ void edge_bias_kernel(const _Float16* __restrict__ ef16,
                                 const float* __restrict__ eW1,
                                 const float* __restrict__ eb1,
                                 const float* __restrict__ eW2,
                                 const float* __restrict__ eb2,
                                 float* __restrict__ attn)
{
  __shared__ float W1[16 * 16], W2[16 * 8], B1[16], B2[8];
  const int t = threadIdx.x;
  if (t < 256) W1[t] = eW1[t];
  if (t < 128) W2[t] = eW2[t];
  if (t < 16)  B1[t] = eb1[t];
  if (t < 8)   B2[t] = eb2[t];
  __syncthreads();

  const size_t idx = (size_t)blockIdx.x * blockDim.x + t;   // pair (i*G + j)
  if (idx >= (size_t)GG_) return;

  const v8h elo = *(const v8h*)&ef16[idx * 16];
  const v8h ehi = *(const v8h*)&ef16[idx * 16 + 8];
  float e[16];
#pragma unroll
  for (int c = 0; c < 8; ++c) { e[c] = (float)elo[c]; e[c + 8] = (float)ehi[c]; }

  float h[16];
#pragma unroll
  for (int j = 0; j < 16; ++j) {
    float s = B1[j];
#pragma unroll
    for (int c = 0; c < 16; ++c) s += e[c] * W1[c * 16 + j];
    h[j] = eluf(s);
  }
#pragma unroll
  for (int hh = 0; hh < 8; ++hh) {
    float s = B2[hh];
#pragma unroll
    for (int j = 0; j < 16; ++j) s += h[j] * W2[j * 8 + hh];
    attn[(size_t)hh * GG_ + idx] += s;
  }
}

// --------------------------------------------------------------------------
// Masked softmax; one wave per (head,row). Fully-masked rows -> zeros
// (nan_to_num). Emits f16 probabilities for the AV GEMM.
// --------------------------------------------------------------------------
__global__ void softmax_kernel(float* __restrict__ attn,
                               const unsigned char* __restrict__ mask,
                               _Float16* __restrict__ p16)
{
  const int row  = blockIdx.x * 8 + (threadIdx.x >> 5);
  const int lane = threadIdx.x & 31;
  if (row >= NH_ * Gn_) return;
  const int i = row % Gn_;
  float* p = attn + (size_t)row * Gn_;
  _Float16* ph = p16 + (size_t)row * Gn_;
  const unsigned char* mrow = mask + (size_t)i * Gn_;

  float mx = -__builtin_inff();
  for (int j = lane; j < Gn_; j += 32) {
    float v = mrow[j] ? -__builtin_inff() : p[j];
    p[j] = v;
    mx = fmaxf(mx, v);
  }
  for (int off = 16; off; off >>= 1) mx = fmaxf(mx, __shfl_xor(mx, off, 32));

  if (mx == -__builtin_inff()) {
    for (int j = lane; j < Gn_; j += 32) { p[j] = 0.f; ph[j] = (_Float16)0.f; }
    return;
  }
  float sum = 0.f;
  for (int j = lane; j < Gn_; j += 32) {
    const float ev = __expf(p[j] - mx);
    p[j] = ev;
    sum += ev;
  }
  for (int off = 16; off; off >>= 1) sum += __shfl_xor(sum, off, 32);
  const float inv = 1.f / sum;
  for (int j = lane; j < Gn_; j += 32) {
    const float v = p[j] * inv;
    p[j] = v;
    ph[j] = (_Float16)v;
  }
}

__global__ void attn_mean_kernel(const float* __restrict__ attn,
                                 float* __restrict__ out)
{
  const size_t idx = (size_t)blockIdx.x * blockDim.x + threadIdx.x;
  if (idx >= (size_t)GG_) return;
  float s = 0.f;
#pragma unroll
  for (int h = 0; h < NH_; ++h) s += attn[(size_t)h * GG_ + idx];
  out[idx] = s * (1.f / NH_);
}

// --------------------------------------------------------------------------
// Fused interaction head: scores = elu(xi@Wi + xj@Wj + ef@We + b1) @ W2 + b2
// without materializing (G,G,256). A = ef16 rows (one b128; K padded to 32),
// B = We^T in LDS (k = hs*8+e, e<8; rest zero). Reduce elu(ph)*W2 in the C layout.
// --------------------------------------------------------------------------
__global__ void interaction_kernel(const _Float16* __restrict__ ef16,
                                   const float* __restrict__ We,   // (16,256)
                                   const float* __restrict__ ai,   // x@Wi f32
                                   const float* __restrict__ bj,   // x@Wj f32
                                   const float* __restrict__ b1,
                                   const float* __restrict__ W2,
                                   const float* __restrict__ b2,
                                   float* __restrict__ scores)
{
  __shared__ __align__(32) _Float16 WeT[HID_][16];
  for (int idx = threadIdx.x; idx < NEF_ * HID_; idx += 128) {
    WeT[idx & 255][idx >> 8] = (_Float16)We[idx];
  }
  __syncthreads();

  const int wave = threadIdx.x >> 5;
  const int lane = threadIdx.x & 31;
  const int m    = lane & 15;
  const int hs   = lane >> 4;
  const long long p0 = ((long long)blockIdx.x * 4 + wave) * 16;
  if (p0 >= (long long)GG_) return;

  // A fragment: ka = hs*8 + e (e<8); e>=8 are K-pad zeros.
  v16h a = {};
  {
    const v8h r = *(const v8h*)&ef16[(size_t)(p0 + m) * NEF_ + (hs << 3)];
#pragma unroll
    for (int e = 0; e < 8; ++e) a[e] = r[e];
  }

  int irow[8], jrow[8];
#pragma unroll
  for (int r = 0; r < 8; ++r) {
    const long long p = p0 + r + (hs << 3);
    irow[r] = (int)(p / Gn_);
    jrow[r] = (int)(p - (long long)irow[r] * Gn_);
  }

  const _Float16 msk = hs ? (_Float16)0.f : (_Float16)1.f;  // unused: hs=1 B lanes hold k=8+e
  float part[8] = {0.f, 0.f, 0.f, 0.f, 0.f, 0.f, 0.f, 0.f};

  for (int ct = 0; ct < HID_ / 16; ++ct) {
    const int col = ct * 16 + m;
    const v8h wlo = *(const v8h*)&WeT[col][hs << 3];
    v16h b = {};
#pragma unroll
    for (int e = 0; e < 8; ++e) { b[e] = wlo[e]; }

    v8f d = {};
    d = __builtin_amdgcn_wmma_f32_16x16x32_f16(
        false, a, false, b, (short)0, d, false, false);

    const float w2c = W2[col];
    const float b1c = b1[col];
#pragma unroll
    for (int r = 0; r < 8; ++r) {
      const float ph = d[r] + ai[(size_t)irow[r] * HID_ + col]
                            + bj[(size_t)jrow[r] * HID_ + col] + b1c;
      part[r] += eluf(ph) * w2c;
    }
  }

#pragma unroll
  for (int r = 0; r < 8; ++r) {
    float s = part[r];
    s += __shfl_xor(s, 1, 32);
    s += __shfl_xor(s, 2, 32);
    s += __shfl_xor(s, 4, 32);
    s += __shfl_xor(s, 8, 32);
    part[r] = s;
  }
  if (m == 0) {
#pragma unroll
    for (int r = 0; r < 8; ++r)
      scores[p0 + r + (hs << 3)] = part[r] + b2[0];
  }
}

// ---------------------------------------------------------------------------
// Host side
// ---------------------------------------------------------------------------
static void launch_gemm(hipStream_t s,
                        const _Float16* A, int lda, long long sAz,
                        const _Float16* Bt, int ldbt, long long sBz,
                        const float* bias, const float* residual,
                        float* C, int ldc, long long sCz,
                        _Float16* C16, long long sC16z,
                        int M, int N, int K, float alpha, int act, int Z)
{
  dim3 grid((N + 127) / 128, M / 16, Z);
  wmma_gemm_f16_kernel<<<grid, 128, 0, s>>>(A, lda, sAz, Bt, ldbt, sBz, bias,
                                            residual, C, ldc, sCz, C16, sC16z,
                                            N, K, alpha, act);
}

static void launch_transpose(hipStream_t s, const float* in, _Float16* out,
                             int R, int C)
{
  transpose_f16_kernel<<<dim3(C / 32, R / 32), 256, 0, s>>>(in, out, R, C);
}

extern "C" void kernel_launch(void* const* d_in, const int* in_sizes, int n_in,
                              void* d_out, int out_size, void* d_ws,
                              size_t ws_size, hipStream_t stream)
{
  (void)in_sizes; (void)n_in; (void)out_size; (void)ws_size;

  // --- inputs (top level in setup_inputs order; params as sorted pytree) ---
  const float*         nf   = (const float*)d_in[0];          // (768,64)
  const float*         ef   = (const float*)d_in[1];          // (768,768,16)
  const unsigned char* mask = (const unsigned char*)d_in[2];  // (768,768) bool
  const float* iW2 = (const float*)d_in[3];
  const float* iWe = (const float*)d_in[4];
  const float* iWi = (const float*)d_in[5];
  const float* iWj = (const float*)d_in[6];
  const float* ib1 = (const float*)d_in[7];
  const float* ib2 = (const float*)d_in[8];
  const float* L[2][20];
  for (int l = 0; l < 2; ++l)
    for (int i = 0; i < 20; ++i)
      L[l][i] = (const float*)d_in[9 + l * 20 + i];
  enum { Wk, Wo, Wq, Wv, bk, bo, bq, bv, eW1, eW2, eb1, eb2,
         fW1, fW2, fb1, fb2, n1b, n1g, n2b, n2g };
  const float* neW  = (const float*)d_in[49];
  const float* neb  = (const float*)d_in[50];
  const float* nebe = (const float*)d_in[51];
  const float* neg_ = (const float*)d_in[52];

  // --- outputs: (x, scores, attn_w) concatenated flat ---
  float* out_x  = (float*)d_out;
  float* out_sc = out_x + (size_t)NX_;
  float* out_aw = out_sc + (size_t)GG_;

  // --- workspace: f32 region then f16 region ---
  float* x    = (float*)d_ws;                    // (768,256)
  float* nb   = x    + NX_;
  float* q    = nb   + NX_;
  float* kbuf = q    + NX_;
  float* v    = kbuf + NX_;
  float* ob   = v    + NX_;
  float* h1   = ob   + NX_;                      // (768,512)
  float* attn = h1   + 2 * NX_;                  // (8,768,768)
  _Float16* ef16 = (_Float16*)(attn + (size_t)NH_ * GG_);
  _Float16* p16  = ef16 + (size_t)GG_ * NEF_;    // f16 attention probs
  _Float16* x16  = p16  + (size_t)NH_ * GG_;
  _Float16* nb16 = x16  + NX_;
  _Float16* q16  = nb16 + NX_;
  _Float16* k16  = q16  + NX_;
  _Float16* ob16 = k16  + NX_;
  _Float16* h116 = ob16 + NX_;                   // (768,512)
  _Float16* vT16 = h116 + 2 * NX_;               // (256,768) = v^T
  _Float16* nf16 = vT16 + NX_;                   // (768,64)
  _Float16* wT   = nf16 + (size_t)Gn_ * NFD_;    // weight^T scratch (<=512x256)

  const float inv_scale = 0.17677669529663687f;  // 1/sqrt(HD=32)

  // --- one-time f16 conversions ---
  cvt_f16_kernel<<<(GG_ * NEF_ / 4 + 255) / 256, 256, 0, stream>>>(
      ef, ef16, GG_ * NEF_ / 4);
  cvt_f16_kernel<<<(Gn_ * NFD_ / 4 + 255) / 256, 256, 0, stream>>>(
      nf, nf16, Gn_ * NFD_ / 4);

  // --- node encoder: x = elu(LN(nf @ W + b)) ---
  launch_transpose(stream, neW, wT, NFD_, HID_);                 // (256,64)
  launch_gemm(stream, nf16, NFD_, 0, wT, NFD_, 0, neb, nullptr,
              nb, HID_, 0, nullptr, 0, Gn_, HID_, NFD_, 1.f, 0, 1);
  layernorm_kernel<<<Gn_, HID_, 0, stream>>>(nb, neg_, nebe, x, x16, HID_, 1);

  // --- transformer layers ---
  for (int l = 0; l < 2; ++l) {
    const float* const* P = L[l];
    layernorm_kernel<<<Gn_, HID_, 0, stream>>>(x, P[n1g], P[n1b], nb, nb16,
                                               HID_, 0);
    launch_transpose(stream, P[Wq], wT, HID_, HID_);
    launch_gemm(stream, nb16, HID_, 0, wT, HID_, 0, P[bq], nullptr,
                q, HID_, 0, q16, 0, Gn_, HID_, HID_, 1.f, 0, 1);
    launch_transpose(stream, P[Wk], wT, HID_, HID_);
    launch_gemm(stream, nb16, HID_, 0, wT, HID_, 0, P[bk], nullptr,
                kbuf, HID_, 0, k16, 0, Gn_, HID_, HID_, 1.f, 0, 1);
    launch_transpose(stream, P[Wv], wT, HID_, HID_);
    launch_gemm(stream, nb16, HID_, 0, wT, HID_, 0, P[bv], nullptr,
                v, HID_, 0, nullptr, 0, Gn_, HID_, HID_, 1.f, 0, 1);
    launch_transpose(stream, v, vT16, Gn_, HID_);                // v^T f16
    // logits: per head, q_h @ k_h^T / sqrt(HD); k16 is already the Bt form
    launch_gemm(stream, q16, HID_, HD_, k16, HID_, HD_, nullptr, nullptr,
                attn, Gn_, (long long)GG_, nullptr, 0,
                Gn_, Gn_, HD_, inv_scale, 0, NH_);
    edge_bias_kernel<<<GG_ / 256, 256, 0, stream>>>(
        ef16, P[eW1], P[eb1], P[eW2], P[eb2], attn);
    softmax_kernel<<<NH_ * Gn_ / 8, 256, 0, stream>>>(attn, mask, p16);
    if (l == 1)
      attn_mean_kernel<<<GG_ / 256, 256, 0, stream>>>(attn, out_aw);
    // out = P @ v per head; Bt = v^T rows [h*32, h*32+32)
    launch_gemm(stream, p16, Gn_, (long long)GG_, vT16, Gn_,
                (long long)HD_ * Gn_, nullptr, nullptr,
                ob, HID_, HD_, ob16, HD_, Gn_, HD_, Gn_, 1.f, 0, NH_);
    // x = x + out @ Wo + bo
    launch_transpose(stream, P[Wo], wT, HID_, HID_);
    launch_gemm(stream, ob16, HID_, 0, wT, HID_, 0, P[bo], x,
                x, HID_, 0, x16, 0, Gn_, HID_, HID_, 1.f, 0, 1);
    // FFN
    layernorm_kernel<<<Gn_, HID_, 0, stream>>>(x, P[n2g], P[n2b], nb, nb16,
                                               HID_, 0);
    launch_transpose(stream, P[fW1], wT, HID_, 2 * HID_);        // (512,256)
    launch_gemm(stream, nb16, HID_, 0, wT, HID_, 0, P[fb1], nullptr,
                h1, 2 * HID_, 0, h116, 0, Gn_, 2 * HID_, HID_, 1.f, 1, 1);
    launch_transpose(stream, P[fW2], wT, 2 * HID_, HID_);        // (256,512)
    launch_gemm(stream, h116, 2 * HID_, 0, wT, 2 * HID_, 0, P[fb2], x,
                x, HID_, 0, x16, 0, Gn_, HID_, 2 * HID_, 1.f, 0, 1);
  }

  // --- interaction head ---
  launch_transpose(stream, iWi, wT, HID_, HID_);
  launch_gemm(stream, x16, HID_, 0, wT, HID_, 0, nullptr, nullptr,
              q, HID_, 0, nullptr, 0, Gn_, HID_, HID_, 1.f, 0, 1);   // ai
  launch_transpose(stream, iWj, wT, HID_, HID_);
  launch_gemm(stream, x16, HID_, 0, wT, HID_, 0, nullptr, nullptr,
              kbuf, HID_, 0, nullptr, 0, Gn_, HID_, HID_, 1.f, 0, 1); // bj
  interaction_kernel<<<dim3(GG_ / 16 / 4), 128, 0, stream>>>(
      ef16, iWe, q, kbuf, ib1, iW2, ib2, out_sc);

  // --- emit x ---
  hipMemcpyAsync(out_x, x, (size_t)NX_ * sizeof(float),
                 hipMemcpyDeviceToDevice, stream);
}
